// MambaBlock_61143154425893
// MI455X (gfx1250) — matrix operand order repaired and checked by reading, not compiled, not run on hardware
//
#include <hip/hip_runtime.h>
#include <math.h>

typedef __attribute__((ext_vector_type(16))) _Float16 v16h;
typedef __attribute__((ext_vector_type(8)))  _Float16 v8h;
typedef __attribute__((ext_vector_type(8)))  float    v8f;

#define D_MODEL 1024
#define D_INNER 2048
#define D_STATE 16
#define D_CONVK 4
#define DT_RANK 64
#define SEQ     2048
#define NTOK    4096                      // B * L
#define XDBL_W  (DT_RANK + 2 * D_STATE)   // 96

__device__ __forceinline__ v16h pack16(v8h lo, v8h hi) {
    return __builtin_shufflevector(lo, hi, 0,1,2,3,4,5,6,7,8,9,10,11,12,13,14,15);
}

// ---------------------------------------------------------------------------
// f32 -> f16 elementwise convert
// ---------------------------------------------------------------------------
__global__ void __launch_bounds__(256)
cvt_f16_kernel(const float* __restrict__ src, _Float16* __restrict__ dst, int n)
{
    int i = blockIdx.x * blockDim.x + threadIdx.x;
    if (i < n) dst[i] = (_Float16)src[i];
}

// ---------------------------------------------------------------------------
// Weight transpose + convert: B[K,N] f32 -> Bt[N,K] f16 (coalesced writes)
// ---------------------------------------------------------------------------
__global__ void __launch_bounds__(256)
transp_cvt_kernel(const float* __restrict__ B, _Float16* __restrict__ Bt, int K, int N)
{
    int i = blockIdx.x * blockDim.x + threadIdx.x;
    if (i >= K * N) return;
    int k = i % K, n = i / K;
    Bt[(size_t)n * K + k] = (_Float16)B[(size_t)k * N + n];
}

// ---------------------------------------------------------------------------
// GEMM: C[M,N] = act(A @ B + bias). A: f16 [M,K] row-major (lda), Bt: f16
// [N,K] (ldb = row stride). One wave computes a 32x32 C block: MT=2 x NT=2
// WMMA tiles, fp32 accumulation. Optional f16 mirror of C into Ch.
// Requires M%32==0, N%32==0, K%32==0 (true for all calls here).
// act: 0 = identity, 1 = softplus
// ---------------------------------------------------------------------------
__global__ void __launch_bounds__(256)
gemm_wmma_f16(const _Float16* __restrict__ A, int lda,
              const _Float16* __restrict__ Bt, int ldb,
              float* __restrict__ C, int ldc,
              _Float16* __restrict__ Ch, int ldch,
              int M, int N, int K,
              const float* __restrict__ bias, int act)
{
    const int lane = threadIdx.x & 31;
    const int wave = threadIdx.x >> 5;
    const int gN   = N >> 5;                       // 32-wide groups in N
    const int g    = blockIdx.x * 8 + wave;        // wave-uniform
    if (g >= (M >> 5) * gN) return;                // EXEC stays all-1s
    const int tm  = (g / gN) << 5;
    const int tn  = (g % gN) << 5;
    const int half = lane >> 4;
    const int mr   = lane & 15;

    v8f acc00 = {}, acc01 = {}, acc10 = {}, acc11 = {};

    const _Float16* Ar0 = A + (size_t)(tm + mr) * lda;
    const _Float16* Ar1 = Ar0 + (size_t)16 * lda;
    const _Float16* Br0 = Bt + (size_t)(tn + mr) * ldb;
    const _Float16* Br1 = Br0 + (size_t)16 * ldb;

    for (int k0 = 0; k0 < K; k0 += 32) {
        const int kA = k0 + 8 * half;        // A: a[0..7]=K[8h..], a[8..15]=K[16+8h..]
        const int kB = k0 + 8 * half;        // B: b[0..7]=K[8h..], b[8..15]=K[16+8h..]
        v16h a0 = pack16(*(const v8h*)(Ar0 + kA), *(const v8h*)(Ar0 + kA + 16));
        v16h a1 = pack16(*(const v8h*)(Ar1 + kA), *(const v8h*)(Ar1 + kA + 16));
        v16h b0 = pack16(*(const v8h*)(Br0 + kB), *(const v8h*)(Br0 + kB + 16));
        v16h b1 = pack16(*(const v8h*)(Br1 + kB), *(const v8h*)(Br1 + kB + 16));
        acc00 = __builtin_amdgcn_wmma_f32_16x16x32_f16(false, a0, false, b0, (short)0, acc00, false, false);
        acc01 = __builtin_amdgcn_wmma_f32_16x16x32_f16(false, a0, false, b1, (short)0, acc01, false, false);
        acc10 = __builtin_amdgcn_wmma_f32_16x16x32_f16(false, a1, false, b0, (short)0, acc10, false, false);
        acc11 = __builtin_amdgcn_wmma_f32_16x16x32_f16(false, a1, false, b1, (short)0, acc11, false, false);
    }

    // C fragment: acc[r] -> row = mTile + 8*half + r, col = nTile + mr
    const float bv0 = bias ? bias[tn + mr]      : 0.0f;
    const float bv1 = bias ? bias[tn + 16 + mr] : 0.0f;
    #pragma unroll
    for (int r = 0; r < 8; ++r) {
        const int r0 = tm + half * 8 + r;
        const int r1 = r0 + 16;
        float v00 = acc00[r] + bv0, v01 = acc01[r] + bv1;
        float v10 = acc10[r] + bv0, v11 = acc11[r] + bv1;
        if (act == 1) {
            v00 = (v00 > 20.0f) ? v00 : log1pf(expf(v00));
            v01 = (v01 > 20.0f) ? v01 : log1pf(expf(v01));
            v10 = (v10 > 20.0f) ? v10 : log1pf(expf(v10));
            v11 = (v11 > 20.0f) ? v11 : log1pf(expf(v11));
        }
        C[(size_t)r0 * ldc + tn + mr]      = v00;
        C[(size_t)r0 * ldc + tn + 16 + mr] = v01;
        C[(size_t)r1 * ldc + tn + mr]      = v10;
        C[(size_t)r1 * ldc + tn + 16 + mr] = v11;
        if (Ch) {
            Ch[(size_t)r0 * ldch + tn + mr]      = (_Float16)v00;
            Ch[(size_t)r0 * ldch + tn + 16 + mr] = (_Float16)v01;
            Ch[(size_t)r1 * ldch + tn + mr]      = (_Float16)v10;
            Ch[(size_t)r1 * ldch + tn + 16 + mr] = (_Float16)v11;
        }
    }
}

// ---------------------------------------------------------------------------
// Causal depthwise conv1d (K=4) + bias + SiLU; writes f32 (for scan) and f16
// (for next GEMM).
// ---------------------------------------------------------------------------
__global__ void __launch_bounds__(256)
conv_silu_kernel(const float* __restrict__ xr,   // (NTOK, 2*D_INNER)
                 const float* __restrict__ cw,   // (D_INNER, 1, 4)
                 const float* __restrict__ cb,
                 float* __restrict__ u,
                 _Float16* __restrict__ u16)
{
    const int idx = blockIdx.x * blockDim.x + threadIdx.x;
    const int c   = idx & (D_INNER - 1);
    const int t   = idx >> 11;
    const int pos = t & (SEQ - 1);
    const int bs  = t - pos;

    float acc = cb[c];
    #pragma unroll
    for (int j = 0; j < D_CONVK; ++j) {
        const int p = pos - (D_CONVK - 1) + j;
        if (p >= 0)
            acc += cw[c * D_CONVK + j] * xr[(size_t)(bs + p) * (2 * D_INNER) + c];
    }
    const float s = acc / (1.0f + expf(-acc));
    u[idx]   = s;
    u16[idx] = (_Float16)s;
}

// ---------------------------------------------------------------------------
// Selective scan: one thread per (batch, channel), 16-state recurrence,
// fused skip + silu(res) gate. Writes f16 directly for the output GEMM.
// ---------------------------------------------------------------------------
__global__ void __launch_bounds__(256)
scan_kernel(const float* __restrict__ delta,
            const float* __restrict__ u,
            const float* __restrict__ xdbl,     // (NTOK, 96): [dlt|B|C]
            const float* __restrict__ xr,       // res = cols [D_INNER, 2*D_INNER)
            const float* __restrict__ A_log,
            const float* __restrict__ Dp,
            _Float16* __restrict__ ypre16)
{
    const int idx   = blockIdx.x * blockDim.x + threadIdx.x;
    const int batch = idx >> 11;
    const int d     = idx & (D_INNER - 1);

    float Ad[D_STATE], h[D_STATE];
    #pragma unroll
    for (int n = 0; n < D_STATE; ++n) {
        Ad[n] = -expf(A_log[d * D_STATE + n]);
        h[n]  = 0.0f;
    }
    const float Dd = Dp[d];

    for (int tl = 0; tl < SEQ; ++tl) {
        const size_t t  = (size_t)batch * SEQ + tl;
        const float dt  = delta[t * D_INNER + d];
        const float ut  = u[t * D_INNER + d];
        const float* xd = xdbl + t * XDBL_W;

        float y = 0.0f;
        #pragma unroll
        for (int n = 0; n < D_STATE; ++n) {
            const float Bn = xd[DT_RANK + n];
            const float Cn = xd[DT_RANK + D_STATE + n];
            h[n] = expf(dt * Ad[n]) * h[n] + dt * Bn * ut;
            y = fmaf(h[n], Cn, y);
        }
        const float res  = xr[t * (2 * D_INNER) + D_INNER + d];
        const float gate = res / (1.0f + expf(-res));
        ypre16[t * D_INNER + d] = (_Float16)((y + ut * Dd) * gate);
    }
}

// ---------------------------------------------------------------------------
extern "C" void kernel_launch(void* const* d_in, const int* in_sizes, int n_in,
                              void* d_out, int out_size, void* d_ws, size_t ws_size,
                              hipStream_t stream)
{
    const float* x     = (const float*)d_in[0];  // (2,2048,1024)
    const float* W_in  = (const float*)d_in[1];  // (1024, 4096)
    const float* convw = (const float*)d_in[2];  // (2048,1,4)
    const float* convb = (const float*)d_in[3];
    const float* W_x   = (const float*)d_in[4];  // (2048, 96)
    const float* W_dt  = (const float*)d_in[5];  // (64, 2048)
    const float* b_dt  = (const float*)d_in[6];
    const float* A_log = (const float*)d_in[7];  // (2048, 16)
    const float* Dp    = (const float*)d_in[8];
    const float* W_out = (const float*)d_in[9];  // (2048, 1024)
    float* out = (float*)d_out;

    // ---- workspace carve-up (256B-aligned slots) ----
    char* base = (char*)d_ws;
    size_t off = 0;
    auto alloc = [&](size_t bytes) -> void* {
        void* p = base + off;
        off += (bytes + 255) & ~(size_t)255;
        return p;
    };
    float*    xr     = (float*)   alloc((size_t)NTOK * 2 * D_INNER * 4);
    float*    u      = (float*)   alloc((size_t)NTOK * D_INNER * 4);
    float*    xdbl   = (float*)   alloc((size_t)NTOK * XDBL_W * 4);
    float*    delta  = (float*)   alloc((size_t)NTOK * D_INNER * 4);
    _Float16* x16    = (_Float16*)alloc((size_t)NTOK * D_MODEL * 2);
    _Float16* u16    = (_Float16*)alloc((size_t)NTOK * D_INNER * 2);
    _Float16* xdbl16 = (_Float16*)alloc((size_t)NTOK * XDBL_W * 2);
    _Float16* ypre16 = (_Float16*)alloc((size_t)NTOK * D_INNER * 2);
    _Float16* Wint   = (_Float16*)alloc((size_t)(2 * D_INNER) * D_MODEL * 2); // (4096,1024)
    _Float16* Wxt    = (_Float16*)alloc((size_t)XDBL_W * D_INNER * 2);        // (96,2048)
    _Float16* Wdtt   = (_Float16*)alloc((size_t)D_INNER * DT_RANK * 2);       // (2048,64)
    _Float16* Woutt  = (_Float16*)alloc((size_t)D_MODEL * D_INNER * 2);       // (1024,2048)

    auto grid1d = [](int n) { return dim3((n + 255) / 256); };

    // ---- precision staging: f16 activations + transposed f16 weights ----
    cvt_f16_kernel<<<grid1d(NTOK * D_MODEL), 256, 0, stream>>>(x, x16, NTOK * D_MODEL);
    transp_cvt_kernel<<<grid1d(D_MODEL * 2 * D_INNER), 256, 0, stream>>>(W_in,  Wint,  D_MODEL, 2 * D_INNER);
    transp_cvt_kernel<<<grid1d(D_INNER * XDBL_W),      256, 0, stream>>>(W_x,   Wxt,   D_INNER, XDBL_W);
    transp_cvt_kernel<<<grid1d(DT_RANK * D_INNER),     256, 0, stream>>>(W_dt,  Wdtt,  DT_RANK, D_INNER);
    transp_cvt_kernel<<<grid1d(D_INNER * D_MODEL),     256, 0, stream>>>(W_out, Woutt, D_INNER, D_MODEL);

    auto gemm = [&](const _Float16* A, int lda, const _Float16* Bt, int ldb,
                    float* C, int ldc, _Float16* Ch, int ldch,
                    int M, int N, int K, const float* bias, int act) {
        const int groups = (M / 32) * (N / 32);
        gemm_wmma_f16<<<dim3((groups + 7) / 8), 256, 0, stream>>>(
            A, lda, Bt, ldb, C, ldc, Ch, ldch, M, N, K, bias, act);
    };

    // 1) xr = x @ W_in                   (4096 x 4096, K=1024)
    gemm(x16, D_MODEL, Wint, D_MODEL, xr, 2 * D_INNER, nullptr, 0,
         NTOK, 2 * D_INNER, D_MODEL, nullptr, 0);

    // 2) u = silu(causal_dwconv(xin) + conv_b)
    conv_silu_kernel<<<grid1d(NTOK * D_INNER), 256, 0, stream>>>(xr, convw, convb, u, u16);

    // 3) x_dbl = u @ W_x                 (4096 x 96, K=2048), + f16 mirror
    gemm(u16, D_INNER, Wxt, D_INNER, xdbl, XDBL_W, xdbl16, XDBL_W,
         NTOK, XDBL_W, D_INNER, nullptr, 0);

    // 4) delta = softplus(dlt @ W_dt + b_dt)  (4096 x 2048, K=64); dlt = xdbl16[:, :64]
    gemm(xdbl16, XDBL_W, Wdtt, DT_RANK, delta, D_INNER, nullptr, 0,
         NTOK, D_INNER, DT_RANK, b_dt, 1);

    // 5) selective scan -> ypre16 = (scan_y + u*D) * silu(res)
    scan_kernel<<<grid1d(NTOK), 256, 0, stream>>>(delta, u, xdbl, xr, A_log, Dp, ypre16);

    // 6) out = ypre @ W_out              (4096 x 1024, K=2048)
    gemm(ypre16, D_INNER, Woutt, D_INNER, out, D_MODEL, nullptr, 0,
         NTOK, D_MODEL, D_INNER, nullptr, 0);
}
